// DotAttention_43430709297648
// MI455X (gfx1250) — matrix operand order repaired and checked by reading, not compiled, not run on hardware
//
#include <hip/hip_runtime.h>
#include <hip/hip_bf16.h>

// Problem constants (from reference): B=32, S=1024, T=512, D=1024, fp32.
#define NB 32
#define NS 1024
#define NT 512
#define ND 1024

#define TILE 128          // output tile (M x N) per workgroup
#define KC   32           // K chunk per WMMA step (bf16 K=32)
#define LDSTR 40          // padded LDS row stride in bf16 elems (80B, 16B aligned)

typedef __attribute__((ext_vector_type(16))) __bf16 v16bf;
typedef __attribute__((ext_vector_type(8)))  __bf16 v8bf;
typedef __attribute__((ext_vector_type(8)))  float  v8f;

// 16 f32 staged in registers (one half-row of a K chunk)
struct Stage16 { float4 q[4]; };

// ---- helpers ---------------------------------------------------------------

__device__ __forceinline__ v8f wmma_bf16(v16bf a, v16bf b, v8f c) {
  // v_wmma_f32_16x16x32_bf16 : D = A(16x32) * B(32x16) + C(16x16 f32)
  return __builtin_amdgcn_wmma_f32_16x16x32_bf16(
      /*neg_a=*/false, a, /*neg_b=*/false, b,
      /*c_mod=*/(short)0, c, /*reuse_a=*/false, /*reuse_b=*/false);
}

__device__ __forceinline__ void load_stage16(Stage16& r, const float* __restrict__ p) {
  const float4* s4 = (const float4*)p;
#pragma unroll
  for (int i = 0; i < 4; ++i) r.q[i] = s4[i];
}

// Split staged f32 into bf16 hi/lo, store row-major into LDS (2x b128 each).
__device__ __forceinline__ void cvt_store16(__bf16* hi, __bf16* lo,
                                            int row, int col, const Stage16& r) {
  __bf16 h[16], l[16];
#pragma unroll
  for (int i = 0; i < 4; ++i) {
    float vv[4] = {r.q[i].x, r.q[i].y, r.q[i].z, r.q[i].w};
#pragma unroll
    for (int j = 0; j < 4; ++j) {
      float x   = vv[j];
      __bf16 hh = (__bf16)x;                 // RNE f32 -> bf16
      h[i*4+j]  = hh;
      l[i*4+j]  = (__bf16)(x - (float)hh);   // residual
    }
  }
  v8bf* dh = (v8bf*)(hi + row * LDSTR + col);
  v8bf* dl = (v8bf*)(lo + row * LDSTR + col);
  dh[0] = *(v8bf*)&h[0]; dh[1] = *(v8bf*)&h[8];
  dl[0] = *(v8bf*)&l[0]; dl[1] = *(v8bf*)&l[8];
}

__device__ __forceinline__ void zero_store16(__bf16* hi, __bf16* lo,
                                             int row, int col) {
  v8bf z;
#pragma unroll
  for (int i = 0; i < 8; ++i) z[i] = (__bf16)0.0f;
  v8bf* dh = (v8bf*)(hi + row * LDSTR + col);
  v8bf* dl = (v8bf*)(lo + row * LDSTR + col);
  dh[0] = z; dh[1] = z; dl[0] = z; dl[1] = z;
}

// A-matrix fragment (16x32 bf16), ISA layout:
// lanes 0-15: row M=lane, K {0..7,16..23}; lanes 16-31: row M=lane-16, K {8..15,24..31}
__device__ __forceinline__ v16bf ld_afrag(const __bf16* base, int row0, int lane) {
  const int m = lane & 15, g = lane >> 4;
  const __bf16* p = base + (row0 + m) * LDSTR + g * 8;
  v8bf x0 = *(const v8bf*)p;
  v8bf x1 = *(const v8bf*)(p + 16);
  v16bf f;
#pragma unroll
  for (int i = 0; i < 8; ++i) { f[i] = x0[i]; f[i + 8] = x1[i]; }
  return f;
}

// B-matrix fragment (32x16 bf16): lane n holds column n, K {0..7,16..23} (lanes 0-15) / {8..15,24..31} (lanes 16-31).
// LDS holds B column-major ([n][k]) so this is 2x b128 reads.
__device__ __forceinline__ v16bf ld_bfrag(const __bf16* base, int col0, int lane) {
  const int n = lane & 15, g = lane >> 4;
  const __bf16* p = base + (col0 + n) * LDSTR + g * 8;
  v8bf x0 = *(const v8bf*)p;
  v8bf x1 = *(const v8bf*)(p + 16);
  v16bf f;
#pragma unroll
  for (int i = 0; i < 8; ++i) { f[i] = x0[i]; f[i + 8] = x1[i]; }
  return f;
}

// Per-wave compute step: 2x4 grid of 16x16 tiles, bf16x3 fp32-emulation.
__device__ __forceinline__ void mma_step(const __bf16* Ah, const __bf16* Al,
                                         const __bf16* Bh, const __bf16* Bl,
                                         int wr, int wc, int lane,
                                         v8f acc[2][4]) {
  v16bf ah[2], al[2], bh[4], bl[4];
#pragma unroll
  for (int i = 0; i < 2; ++i) {
    ah[i] = ld_afrag(Ah, wr + i * 16, lane);
    al[i] = ld_afrag(Al, wr + i * 16, lane);
  }
#pragma unroll
  for (int j = 0; j < 4; ++j) {
    bh[j] = ld_bfrag(Bh, wc + j * 16, lane);
    bl[j] = ld_bfrag(Bl, wc + j * 16, lane);
  }
#pragma unroll
  for (int i = 0; i < 2; ++i)
#pragma unroll
    for (int j = 0; j < 4; ++j) {
      acc[i][j] = wmma_bf16(ah[i], bh[j], acc[i][j]);  // hi*hi
      acc[i][j] = wmma_bf16(ah[i], bl[j], acc[i][j]);  // hi*lo
      acc[i][j] = wmma_bf16(al[i], bh[j], acc[i][j]);  // lo*hi
    }
}

// ---- kernel 1: score[b,t,s] = target[t,b,:] . ctx_masked[b,s,:] ------------
// Writes score (with ==0 -> -inf mask) into attn output region, layout [T,B,S].
// Software pipeline: global loads for chunk k+1 staged in VGPRs while chunk k
// computes; LDS is double-buffered -> one workgroup barrier per chunk.

__global__ __launch_bounds__(256) void score_kernel(
    const float* __restrict__ ctx, const int* __restrict__ lengths,
    const float* __restrict__ tgt, float* __restrict__ attn) {
  __shared__ __bf16 Ah[2][TILE * LDSTR], Al[2][TILE * LDSTR];
  __shared__ __bf16 Bh[2][TILE * LDSTR], Bl[2][TILE * LDSTR];

  const int tid  = threadIdx.x;
  const int lane = tid & 31;
  const int wave = tid >> 5;
  const int b    = blockIdx.z;
  const int t0   = blockIdx.y * TILE;
  const int s0   = blockIdx.x * TILE;
  const int len  = lengths[b];

  // cooperative-load mapping: 2 threads per row, 16 f32 each
  const int lrow = tid >> 1;
  const int lcol = (tid & 1) * 16;
  const float* aptr = tgt + ((size_t)(t0 + lrow) * NB + b) * ND + lcol; // target[t,b,:]
  const float* bptr = ctx + ((size_t)b * NS + (s0 + lrow)) * ND + lcol; // ctx[b,s,:]
  const bool bvalid = (s0 + lrow) < len;

  const int wr = (wave & 3) * 32;   // wave's row offset within tile
  const int wc = (wave >> 2) * 64;  // wave's col offset within tile

  v8f acc[2][4] = {};

  Stage16 sa, sb;
  load_stage16(sa, aptr);           // prologue: stage chunk 0
  load_stage16(sb, bptr);

  for (int k0 = 0; k0 < ND; k0 += KC) {
    const int buf = (k0 >> 5) & 1;
    // commit staged chunk k0 into LDS[buf] (other waves may still be computing
    // chunk k0-1 out of LDS[buf^1] -- safe).
    cvt_store16(Ah[buf], Al[buf], lrow, lcol, sa);
    if (bvalid) cvt_store16(Bh[buf], Bl[buf], lrow, lcol, sb);
    else        zero_store16(Bh[buf], Bl[buf], lrow, lcol);  // pad_packed_sequence
    __syncthreads();
    if (k0 + KC < ND) {             // stage chunk k0+1; latency hides under WMMA
      load_stage16(sa, aptr + k0 + KC);
      load_stage16(sb, bptr + k0 + KC);
      if (k0 + 2 * KC < ND) {
        __builtin_prefetch(aptr + k0 + 2 * KC, 0, 3);
        __builtin_prefetch(bptr + k0 + 2 * KC, 0, 3);
      }
    }
    mma_step(Ah[buf], Al[buf], Bh[buf], Bl[buf], wr, wc, lane, acc);
  }

  // C/D layout: lanes 0-15 -> N=lane, M=r; lanes 16-31 -> N=lane-16, M=8+r
  const int n  = lane & 15;
  const int m0 = (lane >> 4) * 8;
  const float NEG_INF = -__builtin_inff();
#pragma unroll
  for (int i = 0; i < 2; ++i)
#pragma unroll
    for (int j = 0; j < 4; ++j) {
      const int trow = t0 + wr + i * 16 + m0;
      const int scol = s0 + wc + j * 16 + n;
#pragma unroll
      for (int r = 0; r < 8; ++r) {
        float v = acc[i][j][r];
        // masked_fill_(score == 0, -inf): padded columns dot to exactly 0
        attn[((size_t)(trow + r) * NB + b) * NS + scol] = (v == 0.0f) ? NEG_INF : v;
      }
    }
}

// ---- kernel 2: row softmax over S (in place, rows are contiguous 4KB) ------

__global__ __launch_bounds__(256) void softmax_kernel(float* __restrict__ attn) {
  __shared__ float red[256];
  const int row = blockIdx.x;           // 0 .. T*B-1
  const int tid = threadIdx.x;
  float4* p = (float4*)(attn + (size_t)row * NS);
  float4 v = p[tid];                     // 256 threads * 4 = S
  float m = fmaxf(fmaxf(v.x, v.y), fmaxf(v.z, v.w));
  red[tid] = m;
  __syncthreads();
  for (int s = 128; s > 0; s >>= 1) {
    if (tid < s) red[tid] = fmaxf(red[tid], red[tid + s]);
    __syncthreads();
  }
  const float mx = red[0];
  __syncthreads();
  const float e0 = __expf(v.x - mx), e1 = __expf(v.y - mx),
              e2 = __expf(v.z - mx), e3 = __expf(v.w - mx);
  red[tid] = e0 + e1 + e2 + e3;
  __syncthreads();
  for (int s = 128; s > 0; s >>= 1) {
    if (tid < s) red[tid] += red[tid + s];
    __syncthreads();
  }
  const float inv = 1.0f / red[0];
  float4 o; o.x = e0 * inv; o.y = e1 * inv; o.z = e2 * inv; o.w = e3 * inv;
  p[tid] = o;
}

// ---- kernel 3: result[t,b,d] = sum_s attn[t,b,s] * ctx[b,s,d] --------------
// Padded s have attn == 0 after softmax, so no masking needed here.

__global__ __launch_bounds__(256) void av_kernel(
    const float* __restrict__ ctx, const float* __restrict__ attn,
    float* __restrict__ outr) {
  __shared__ __bf16 Ah[2][TILE * LDSTR], Al[2][TILE * LDSTR];
  __shared__ __bf16 Bh[2][TILE * LDSTR], Bl[2][TILE * LDSTR];

  const int tid  = threadIdx.x;
  const int lane = tid & 31;
  const int wave = tid >> 5;
  const int b    = blockIdx.z;
  const int t0   = blockIdx.y * TILE;
  const int d0   = blockIdx.x * TILE;

  // A = attn rows (K = S)
  const int arow = tid >> 1;
  const int acol = (tid & 1) * 16;
  const float* aptr = attn + ((size_t)(t0 + arow) * NB + b) * NS + acol;

  // B = ctx[b] (S x D); transpose-store into LDS so K is contiguous per column
  const int bsrow = tid >> 3;          // k within chunk: 0..31
  const int bdcol = (tid & 7) * 16;    // d offset in tile: 0..112
  const float* bptr = ctx + ((size_t)b * NS + bsrow) * ND + d0 + bdcol;

  const int wr = (wave & 3) * 32;
  const int wc = (wave >> 2) * 64;

  v8f acc[2][4] = {};

  Stage16 sa, sb;
  load_stage16(sa, aptr);
  load_stage16(sb, bptr);

  for (int k0 = 0; k0 < NS; k0 += KC) {
    const int buf = (k0 >> 5) & 1;
    cvt_store16(Ah[buf], Al[buf], arow, acol, sa);
    {
      // transpose-store ctx chunk: LDS B is [d][k]
      __bf16* __restrict__ bh = Bh[buf];
      __bf16* __restrict__ bl = Bl[buf];
#pragma unroll
      for (int i = 0; i < 4; ++i) {
        float vv[4] = {sb.q[i].x, sb.q[i].y, sb.q[i].z, sb.q[i].w};
#pragma unroll
        for (int j = 0; j < 4; ++j) {
          float x   = vv[j];
          __bf16 hh = (__bf16)x;
          bh[(bdcol + i * 4 + j) * LDSTR + bsrow] = hh;
          bl[(bdcol + i * 4 + j) * LDSTR + bsrow] = (__bf16)(x - (float)hh);
        }
      }
    }
    __syncthreads();
    if (k0 + KC < NS) {
      load_stage16(sa, aptr + k0 + KC);
      load_stage16(sb, bptr + (size_t)(k0 + KC) * ND);
      if (k0 + 2 * KC < NS) {
        __builtin_prefetch(aptr + k0 + 2 * KC, 0, 3);
        __builtin_prefetch(bptr + (size_t)(k0 + 2 * KC) * ND, 0, 3);
      }
    }
    mma_step(Ah[buf], Al[buf], Bh[buf], Bl[buf], wr, wc, lane, acc);
  }

  const int n  = lane & 15;
  const int m0 = (lane >> 4) * 8;
#pragma unroll
  for (int i = 0; i < 2; ++i)
#pragma unroll
    for (int j = 0; j < 4; ++j) {
      const int trow = t0 + wr + i * 16 + m0;
      const int dcol = d0 + wc + j * 16 + n;
#pragma unroll
      for (int r = 0; r < 8; ++r)
        outr[((size_t)(trow + r) * NB + b) * ND + dcol] = acc[i][j][r];
    }
}

// ---- launcher --------------------------------------------------------------

extern "C" void kernel_launch(void* const* d_in, const int* in_sizes, int n_in,
                              void* d_out, int out_size, void* d_ws, size_t ws_size,
                              hipStream_t stream) {
  (void)in_sizes; (void)n_in; (void)out_size; (void)d_ws; (void)ws_size;
  const float* ctx     = (const float*)d_in[0];  // [B,S,D] f32
  const int*   lengths = (const int*)d_in[1];    // [B] i32
  const float* tgt     = (const float*)d_in[2];  // [T,B,D] f32

  float* attn   = (float*)d_out;                     // [T,B,S]
  float* result = attn + (size_t)NT * NB * NS;       // [T,B,D]

  dim3 blk(256);
  score_kernel<<<dim3(NS / TILE, NT / TILE, NB), blk, 0, stream>>>(ctx, lengths, tgt, attn);
  softmax_kernel<<<dim3(NT * NB), blk, 0, stream>>>(attn);
  av_kernel<<<dim3(ND / TILE, NT / TILE, NB), blk, 0, stream>>>(ctx, attn, result);
}
